// SelfAttention_16166256902615
// MI455X (gfx1250) — matrix operand order repaired and checked by reading, not compiled, not run on hardware
//
#include <hip/hip_runtime.h>

// MI455X / gfx1250 self-attention, f16 WMMA (16x16x32) with fp32 accumulate.
// Pipeline: transpose/cast -> weight casts -> QKV WMMA GEMM -> transposed
// flash-attention (K/V tiles staged to LDS with gfx1250 async global->LDS
// copies, double buffered; softmax rows = lane%16; P^T B-operand built with
// lane-half shuffles) -> output-projection WMMA GEMM.

typedef __attribute__((ext_vector_type(16))) _Float16 v16h;
typedef __attribute__((ext_vector_type(8)))  _Float16 v8h;
typedef __attribute__((ext_vector_type(8)))  float    v8f;
typedef __attribute__((ext_vector_type(4)))  float    v4f;

#define BB 4
#define DD 512
#define HH 8
#define DV 64
#define SS 2048

// workspace layout in _Float16 elements
#define XT_OFF   0u          // (B,S,D)          4194304
#define WQKV_OFF 4194304u    // (3,H,DV,D)        786432
#define W0H_OFF  4980736u    // (D,D)             262144
#define Q_OFF    5242880u    // (B,H,S,DV)       4194304
#define K_OFF    9437184u    // (B,H,S,DV)       4194304
#define VT_OFF   13631488u   // (B,H,DV,S)       4194304
#define HD_OFF   17825792u   // (B,S,D)          4194304

// Build a v16h A/B operand from two contiguous 8-halfword chunks.
__device__ __forceinline__ v16h load_ab(const _Float16* p0, const _Float16* p1) {
  v8h lo = *(const v8h*)p0;
  v8h hi = *(const v8h*)p1;
  v16h r;
#pragma unroll
  for (int i = 0; i < 8; ++i) { r[i] = lo[i]; r[8 + i] = hi[i]; }
  return r;
}

__device__ __forceinline__ v8f wmma16(v16h a, v16h b, v8f c) {
  return __builtin_amdgcn_wmma_f32_16x16x32_f16(false, a, false, b, (short)0, c,
                                                false, false);
}

// gfx1250 async global->LDS copy, GVS addressing: SGPR base + 32-bit VGPR
// byte offset; VDST VGPR carries the per-lane LDS byte address. ASYNCcnt.
__device__ __forceinline__ void async_copy_b128(unsigned lds_byte_addr,
                                                unsigned global_byte_off,
                                                const _Float16* base) {
  asm volatile("global_load_async_to_lds_b128 %0, %1, %2"
               :
               : "v"(lds_byte_addr), "v"(global_byte_off),
                 "s"((unsigned long long)base)
               : "memory");
}

// ---------------- kernel 1: x (B,D,S) f32 -> xt (B,S,D) f16 -----------------
__global__ void xpose_cast(const float* __restrict__ x, _Float16* __restrict__ xt) {
  __shared__ float tile[32][33];
  int bid = blockIdx.x;
  int st = bid % (SS / 32); bid /= (SS / 32);
  int dt = bid % (DD / 32);
  int b  = bid / (DD / 32);
  int tx = threadIdx.x & 31, ty = threadIdx.x >> 5;  // ty 0..7
  int s0 = st * 32, d0 = dt * 32;
#pragma unroll
  for (int k = 0; k < 4; ++k) {
    int d = d0 + ty + 8 * k;
    tile[ty + 8 * k][tx] = x[((size_t)b * DD + d) * SS + s0 + tx];
  }
  __syncthreads();
#pragma unroll
  for (int k = 0; k < 4; ++k) {
    int s = s0 + ty + 8 * k;
    xt[((size_t)b * SS + s) * DD + d0 + tx] = (_Float16)tile[tx][ty + 8 * k];
  }
}

// ---------------- kernel 2: f32 -> f16 cast ---------------------------------
__global__ void cast_f32_f16(const float* __restrict__ src,
                             _Float16* __restrict__ dst, int n) {
  int i = blockIdx.x * 256 + threadIdx.x;
  if (i < n) dst[i] = (_Float16)src[i];
}

// ---------------- kernel 3: QKV projection ----------------------------------
// Transposed GEMM: C(e,s) = W(e,d) x xt^T(d,s), K-loop over D in steps of 32.
// Q,K stored (B,H,S,DV); V stored transposed (B,H,DV,S).
__global__ void qkv_proj(const _Float16* __restrict__ xt,
                         const _Float16* __restrict__ w,
                         const float* __restrict__ bq,
                         const float* __restrict__ bk,
                         const float* __restrict__ bv,
                         _Float16* __restrict__ Q, _Float16* __restrict__ K,
                         _Float16* __restrict__ VT) {
  int wid = blockIdx.x * (blockDim.x >> 5) + (threadIdx.x >> 5);
  int lane = threadIdx.x & 31, half = lane >> 4, l16 = lane & 15;
  int st = wid % (SS / 16); wid /= (SS / 16);
  int et = wid % (DV / 16); wid /= (DV / 16);
  int h  = wid % HH;        wid /= HH;
  int mu = wid % 3;
  int b  = wid / 3;
  int s0 = st * 16, e0 = et * 16;

  const _Float16* wrow = w  + ((size_t)(mu * HH + h) * DV + e0 + l16) * DD;
  const _Float16* xrow = xt + ((size_t)b * SS + s0 + l16) * DD;

  v8f acc = {};
#pragma unroll 4
  for (int c = 0; c < DD / 32; ++c) {
    int d = c * 32;
    v16h a  = load_ab(wrow + d + half * 8, wrow + d + 16 + half * 8);
    v16h bo = load_ab(xrow + d + half * 8, xrow + d + 16 + half * 8);
    acc = wmma16(a, bo, acc);
  }

  const float* bias =
      (mu == 0 ? bq : (mu == 1 ? bk : bv)) + h * DV + e0 + half * 8;
  v4f b0v = *(const v4f*)bias;
  v4f b1v = *(const v4f*)(bias + 4);
  float bias8[8];
#pragma unroll
  for (int i = 0; i < 4; ++i) { bias8[i] = b0v[i]; bias8[4 + i] = b1v[i]; }

  size_t bh = (size_t)(b * HH + h);
  if (mu < 2) {
    // element r: e = e0+half*8+r (contiguous), s = s0+l16 fixed -> one v8h store
    _Float16* dst =
        (mu == 0 ? Q : K) + (bh * SS + s0 + l16) * DV + e0 + half * 8;
    v8h o;
#pragma unroll
    for (int r = 0; r < 8; ++r) o[r] = (_Float16)(acc[r] + bias8[r]);
    *(v8h*)dst = o;
  } else {
#pragma unroll
    for (int r = 0; r < 8; ++r) {
      VT[(bh * DV + e0 + half * 8 + r) * SS + s0 + l16] =
          (_Float16)(acc[r] + bias8[r]);
    }
  }
}

// ---------------- kernel 4: flash attention (transposed, LDS-staged) --------
// One block (8 waves) per (b,h,128-row s group); all waves share K/V tiles
// staged in LDS via async global->LDS copies, double-buffered.
__global__ void attn(const _Float16* __restrict__ Q,
                     const _Float16* __restrict__ K,
                     const _Float16* __restrict__ VT,
                     const float* __restrict__ mask,
                     _Float16* __restrict__ heads) {
  __shared__ __align__(16) _Float16 kbuf[2][32][DV];  // [t_local][e]
  __shared__ __align__(16) _Float16 vbuf[2][DV][32];  // [e][t_local]

  int bid = blockIdx.x;
  int sg  = bid % (SS / 128);      // s group of 128 rows
  int bhi = bid / (SS / 128);
  int h   = bhi % HH;
  int b   = bhi / HH;
  int tid = threadIdx.x;
  int wv  = tid >> 5;
  int lane = tid & 31, half = lane >> 4, l16 = lane & 15;
  int s0 = sg * 128 + wv * 16;
  size_t bh = (size_t)(b * HH + h);

  const _Float16* kbase = K  + bh * SS * DV;   // uniform across block
  const _Float16* vbase = VT + bh * DV * SS;   // uniform across block

  // per-thread async-copy slots: one b128 of K, one b128 of V per tile
  int krow = tid >> 3, kcol = (tid & 7) * 8;   // 32 rows x 64 halfs
  int vrow = tid >> 2, vcol = (tid & 3) * 8;   // 64 rows x 32 halfs

  // prologue: stage tile for tb = 0 into buffer 0
  async_copy_b128((unsigned)(size_t)&kbuf[0][krow][kcol],
                  (unsigned)((krow * DV + kcol) * 2), kbase);
  async_copy_b128((unsigned)(size_t)&vbuf[0][vrow][vcol],
                  (unsigned)((vrow * SS + vcol) * 2), vbase);

  // Q as B-operand (K-dim = e): loaded once from global.
  const _Float16* qrow = Q + (bh * SS + s0 + l16) * DV;
  v16h bq0 = load_ab(qrow + half * 8, qrow + 16 + half * 8);
  v16h bq1 = load_ab(qrow + 32 + half * 8, qrow + 48 + half * 8);

  const float* mrow = mask + (size_t)b * SS;
  float m_i = -3.0e38f, l_i = 0.0f;
  v8f o0 = {}, o1 = {}, o2 = {}, o3 = {};

  for (int tb = 0; tb < SS; tb += 32) {
    int cur = (tb >> 5) & 1;
    __syncthreads();  // everyone finished reading buf[cur^1] (previous tile)
    if (tb + 32 < SS) {
      int nxt = cur ^ 1;
      async_copy_b128((unsigned)(size_t)&kbuf[nxt][krow][kcol],
                      (unsigned)(((tb + 32 + krow) * DV + kcol) * 2), kbase);
      async_copy_b128((unsigned)(size_t)&vbuf[nxt][vrow][vcol],
                      (unsigned)((vrow * SS + tb + 32 + vcol) * 2), vbase);
      asm volatile("s_wait_asynccnt 0x2" ::: "memory");  // current tile landed
    } else {
      asm volatile("s_wait_asynccnt 0x0" ::: "memory");
    }
    __syncthreads();  // all waves' copies for buf[cur] visible

    // scores: S^T = K Q^T ; two 16x16 tiles (t lo/hi), K-dim e=64 -> 2 WMMAs ea
    const _Float16* k0 = &kbuf[cur][l16][0];
    const _Float16* k1 = &kbuf[cur][16 + l16][0];
    v8f cs0 = {}, cs1 = {};
    cs0 = wmma16(load_ab(k0 + half * 8, k0 + 16 + half * 8), bq0, cs0);
    cs0 = wmma16(load_ab(k0 + 32 + half * 8, k0 + 48 + half * 8), bq1, cs0);
    cs1 = wmma16(load_ab(k1 + half * 8, k1 + 16 + half * 8), bq0, cs1);
    cs1 = wmma16(load_ab(k1 + 32 + half * 8, k1 + 48 + half * 8), bq1, cs1);

    // scale + key mask; tile0 element r has t = tb + half*8 + r, tile1 +16
    v4f ma0 = *(const v4f*)(mrow + tb + half * 8);
    v4f ma1 = *(const v4f*)(mrow + tb + half * 8 + 4);
    v4f mb0 = *(const v4f*)(mrow + tb + 16 + half * 8);
    v4f mb1 = *(const v4f*)(mrow + tb + 16 + half * 8 + 4);
    float sv0[8], sv1[8];
    float bm = -3.0e38f;
#pragma unroll
    for (int r = 0; r < 8; ++r) {
      float mva = (r < 4) ? ma0[r] : ma1[r - 4];
      float mvb = (r < 4) ? mb0[r] : mb1[r - 4];
      float a = cs0[r] * 0.125f;
      float c = cs1[r] * 0.125f;
      a = a * mva + (1.0f - mva) * (-1.0e30f);
      c = c * mvb + (1.0f - mvb) * (-1.0e30f);
      sv0[r] = a; sv1[r] = c;
      bm = fmaxf(bm, fmaxf(a, c));
    }
    bm = fmaxf(bm, __shfl_xor(bm, 16, 32));  // combine lane halves
    float mn = fmaxf(m_i, bm);
    float alpha = __expf(m_i - mn);
    float p0[8], p1[8];
    float bs = 0.0f;
#pragma unroll
    for (int r = 0; r < 8; ++r) {
      p0[r] = __expf(sv0[r] - mn);
      p1[r] = __expf(sv1[r] - mn);
      bs += p0[r] + p1[r];
    }
    bs += __shfl_xor(bs, 16, 32);
    l_i = l_i * alpha + bs;
    m_i = mn;
#pragma unroll
    for (int r = 0; r < 8; ++r) {
      o0[r] *= alpha; o1[r] *= alpha; o2[r] *= alpha; o3[r] *= alpha;
    }

    // Build P^T B-operand (K-dim = t = half*16 + j) via lane-half exchange.
    v16h bp;
#pragma unroll
    for (int i = 0; i < 8; ++i) {
      float sw0 = __shfl_xor(p0[i], 16, 32);
      float sw1 = __shfl_xor(p1[i], 16, 32);
      bp[i]     = (_Float16)((half == 0) ? p0[i] : sw1);
      bp[8 + i] = (_Float16)((half == 0) ? sw0 : p1[i]);
    }

    // O^T += V^T x P^T ; A = V^T rows (e) from LDS, K-dim = t
    const _Float16* v0 = &vbuf[cur][0  + l16][0];
    const _Float16* v1 = &vbuf[cur][16 + l16][0];
    const _Float16* v2 = &vbuf[cur][32 + l16][0];
    const _Float16* v3 = &vbuf[cur][48 + l16][0];
    o0 = wmma16(load_ab(v0 + half * 16, v0 + half * 16 + 8), bp, o0);
    o1 = wmma16(load_ab(v1 + half * 16, v1 + half * 16 + 8), bp, o1);
    o2 = wmma16(load_ab(v2 + half * 16, v2 + half * 16 + 8), bp, o2);
    o3 = wmma16(load_ab(v3 + half * 16, v3 + half * 16 + 8), bp, o3);
  }

  // epilogue: 1/l, query mask, store heads (B,S,D) f16; e contiguous per tile
  float qm = mrow[s0 + l16];
  float scale = qm / l_i;
  _Float16* hrow = heads + ((size_t)b * SS + s0 + l16) * DD + h * DV + half * 8;
  v8h w0s, w1s, w2s, w3s;
#pragma unroll
  for (int r = 0; r < 8; ++r) {
    w0s[r] = (_Float16)(o0[r] * scale);
    w1s[r] = (_Float16)(o1[r] * scale);
    w2s[r] = (_Float16)(o2[r] * scale);
    w3s[r] = (_Float16)(o3[r] * scale);
  }
  *(v8h*)(hrow +  0) = w0s;
  *(v8h*)(hrow + 16) = w1s;
  *(v8h*)(hrow + 32) = w2s;
  *(v8h*)(hrow + 48) = w3s;
}

// ---------------- kernel 5: output projection -------------------------------
// y^T(n,s) = W0(n,d) x heads^T(d,s) + b0; store out (B,D,S) f32.
__global__ void out_proj(const _Float16* __restrict__ heads,
                         const _Float16* __restrict__ w0,
                         const float* __restrict__ b0,
                         float* __restrict__ out) {
  int wid = blockIdx.x * (blockDim.x >> 5) + (threadIdx.x >> 5);
  int lane = threadIdx.x & 31, half = lane >> 4, l16 = lane & 15;
  int st = wid % (SS / 16); wid /= (SS / 16);
  int nt = wid % (DD / 16);
  int b  = wid / (DD / 16);
  int s0 = st * 16, n0 = nt * 16;

  const _Float16* wrow = w0 + (size_t)(n0 + l16) * DD;
  const _Float16* hrow = heads + ((size_t)b * SS + s0 + l16) * DD;
  v8f acc = {};
#pragma unroll 4
  for (int c = 0; c < DD / 32; ++c) {
    int d = c * 32;
    v16h a  = load_ab(wrow + d + half * 8, wrow + d + 16 + half * 8);
    v16h bo = load_ab(hrow + d + half * 8, hrow + d + 16 + half * 8);
    acc = wmma16(a, bo, acc);
  }
  const float* bp = b0 + n0 + half * 8;
  v4f b0v = *(const v4f*)bp;
  v4f b1v = *(const v4f*)(bp + 4);
#pragma unroll
  for (int r = 0; r < 8; ++r) {
    int n = n0 + half * 8 + r;
    float bias = (r < 4) ? b0v[r] : b1v[r - 4];
    out[((size_t)b * DD + n) * SS + s0 + l16] = acc[r] + bias;
  }
}

// ---------------- host launch ------------------------------------------------
extern "C" void kernel_launch(void* const* d_in, const int* in_sizes, int n_in,
                              void* d_out, int out_size, void* d_ws,
                              size_t ws_size, hipStream_t stream) {
  const float* x    = (const float*)d_in[0];
  const float* mask = (const float*)d_in[1];
  const float* Wq   = (const float*)d_in[2];
  const float* bq   = (const float*)d_in[3];
  const float* Wk   = (const float*)d_in[4];
  const float* bk   = (const float*)d_in[5];
  const float* Wv   = (const float*)d_in[6];
  const float* bv   = (const float*)d_in[7];
  const float* W0   = (const float*)d_in[8];
  const float* b0   = (const float*)d_in[9];
  float* out = (float*)d_out;

  _Float16* ws   = (_Float16*)d_ws;
  _Float16* XT   = ws + XT_OFF;
  _Float16* WQKV = ws + WQKV_OFF;
  _Float16* W0H  = ws + W0H_OFF;
  _Float16* Qb   = ws + Q_OFF;
  _Float16* Kb   = ws + K_OFF;
  _Float16* VT   = ws + VT_OFF;
  _Float16* HD   = ws + HD_OFF;

  const int NW = HH * DV * DD;  // 262144 per weight tensor

  xpose_cast<<<BB * (DD / 32) * (SS / 32), 256, 0, stream>>>(x, XT);
  cast_f32_f16<<<(NW + 255) / 256, 256, 0, stream>>>(Wq, WQKV, NW);
  cast_f32_f16<<<(NW + 255) / 256, 256, 0, stream>>>(Wk, WQKV + NW, NW);
  cast_f32_f16<<<(NW + 255) / 256, 256, 0, stream>>>(Wv, WQKV + 2 * NW, NW);
  cast_f32_f16<<<(NW + 255) / 256, 256, 0, stream>>>(W0, W0H, DD * DD);

  // tiles: 4 * 3 * 8 * 4 * 128 = 49152 waves / 8 per block
  qkv_proj<<<49152 / 8, 256, 0, stream>>>(XT, WQKV, bq, bk, bv, Qb, Kb, VT);
  // blocks: (b,h) x s-groups of 128 = 32 * 16 = 512, 8 waves each
  attn<<<BB * HH * (SS / 128), 256, 0, stream>>>(Qb, Kb, VT, mask, HD);
  // tiles: 4 * 32 * 128 = 16384 waves / 8 per block
  out_proj<<<16384 / 8, 256, 0, stream>>>(HD, W0H, b0, out);
}
